// PosMLP_12232066859090
// MI455X (gfx1250) — matrix operand order repaired and checked by reading, not compiled, not run on hardware
//
#include <hip/hip_runtime.h>
#include <hip/hip_bf16.h>

#define VOCAB   32000
#define MAX_LEN 64
#define HIDDEN  96
#define BATCH   4096

typedef __attribute__((ext_vector_type(2)))  float    v2f;
typedef __attribute__((ext_vector_type(8)))  float    v8f;
typedef __attribute__((ext_vector_type(16))) _Float16 v16h;

// ---------------------------------------------------------------------------
// Kernel 1: embedding-bag gather + reduce + bias + ReLU  (bandwidth bound)
// One block per batch row. Thread t owns hidden channel t (0..95), so each
// position's 96-float row is one coalesced 384B (3x128B-line) fetch.
// Full unroll -> up to 64 independent global loads in flight per lane.
// ---------------------------------------------------------------------------
__global__ __launch_bounds__(HIDDEN) void posmlp_gather(
    const int* __restrict__ ids, const float* __restrict__ W1,
    const float* __restrict__ b1, float* __restrict__ h) {
  __shared__ int sid[MAX_LEN];
  const int b = blockIdx.x;
  const int t = threadIdx.x;
  if (t < MAX_LEN) sid[t] = ids[b * MAX_LEN + t];
  __syncthreads();

  float acc = 0.0f;
#pragma unroll
  for (int pos = 0; pos < MAX_LEN; ++pos) {
    // max index = 64*32000*96 = 196.6M < 2^31 -> 32-bit math is safe
    unsigned off = (unsigned)pos * (unsigned)(VOCAB * HIDDEN) +
                   (unsigned)sid[pos] * (unsigned)HIDDEN + (unsigned)t;
    acc += W1[off];
  }
  float v = acc + b1[t];
  h[b * HIDDEN + t] = v > 0.0f ? v : 0.0f;
}

// ---------------------------------------------------------------------------
// Kernel 2: out[4096,2] = h[4096,96] @ W2[96,2] + b2  via CDNA5 WMMA.
// One wave per 16-row tile (256 tiles). W2 zero-padded to 16 columns is
// staged in LDS once per block (branch-free), so the WMMA loop has no
// EXEC-mask toggling: global_load_b64 (A) + 2x ds_load (B) + v_wmma.
// ---------------------------------------------------------------------------
__global__ __launch_bounds__(256) void posmlp_proj_wmma(
    const float* __restrict__ h, const float* __restrict__ W2,
    const float* __restrict__ b2, float* __restrict__ out) {
  __shared__ float Bs[HIDDEN * 16];  // [k][n], n padded 2 -> 16 with zeros

  // Branch-free padded fill: 1536 elements / 256 threads = 6 iterations.
#pragma unroll
  for (int i = threadIdx.x; i < HIDDEN * 16; i += 256) {
    const int k = i >> 4, n = i & 15;
    // clamped read + data-select (v_cndmask), no divergent load
    const float w = W2[(k * 2 + (n & 1)) ];
    Bs[i] = (n < 2) ? w : 0.0f;
  }
  __syncthreads();

  const int wave = (blockIdx.x * blockDim.x + threadIdx.x) >> 5; // tile 0..255
  const int lane = threadIdx.x & 31;
  const int half = lane >> 4;   // 0: lanes 0-15, 1: lanes 16-31
  const int m    = lane & 15;   // row within tile (A) / column N (B,D)
  const int row  = wave * 16 + m;

  v8f acc = {};

#if __has_builtin(__builtin_amdgcn_wmma_f32_16x16x4_f32)
  // Exact f32 path: 24 x V_WMMA_F32_16X16X4_F32 over K=96.
  // A 16x4 layout (ISA 7.12.2): lane half -> K base {0,2}; VGPR v -> K+v.
  // B 4x16: row K striped across lanes within a VGPR (mirror of C/D).
#pragma unroll
  for (int kk = 0; kk < HIDDEN / 4; ++kk) {
    const int k = kk * 4 + half * 2;
    v2f a, bm;
    a.x  = h[row * HIDDEN + k + 0];
    a.y  = h[row * HIDDEN + k + 1];
    bm.x = Bs[(k + 0) * 16 + m];   // conflict-free: lanes hit distinct banks
    bm.y = Bs[(k + 1) * 16 + m];
    acc = __builtin_amdgcn_wmma_f32_16x16x4_f32(
        /*neg_a=*/false, a, /*neg_b=*/false, bm,
        /*c_mod=*/(short)0, acc, /*reuse_a=*/false, /*reuse_b=*/false);
  }
#else
  // Fallback (codegen-confirmed builtin): 3 x V_WMMA_F32_16X16X32_F16.
#pragma unroll
  for (int chunk = 0; chunk < HIDDEN / 32; ++chunk) {
    v16h a, bm;
#pragma unroll
    for (int e = 0; e < 16; ++e) {
      const int p = e >> 1, q = e & 1;
      const int ka = chunk * 32 + ((p < 4) ? 0 : 16) + (p & 3) * 2 + q + half * 8;
      a[e] = (_Float16)h[row * HIDDEN + ka];
      const int kb = chunk * 32 + ((e < 8) ? e : e + 8) + half * 8;
      bm[e] = (_Float16)Bs[kb * 16 + m];
    }
    acc = __builtin_amdgcn_wmma_f32_16x16x32_f16(
        false, a, false, bm, (short)0, acc, false, false);
  }
#endif

  // D layout: VGPR r, lane (half,m) -> element (M = r + half*8, N = m).
  if (m < 2) {
    const float bias = b2[m];
#pragma unroll
    for (int r = 0; r < 8; ++r) {
      const int M = r + half * 8;
      out[(wave * 16 + M) * 2 + m] = acc[r] + bias;
    }
  }
}

extern "C" void kernel_launch(void* const* d_in, const int* in_sizes, int n_in,
                              void* d_out, int out_size, void* d_ws, size_t ws_size,
                              hipStream_t stream) {
  const int*   ids = (const int*)d_in[0];
  const float* W1  = (const float*)d_in[1];
  const float* b1  = (const float*)d_in[2];
  const float* W2  = (const float*)d_in[3];
  const float* b2  = (const float*)d_in[4];
  float* out = (float*)d_out;
  float* h   = (float*)d_ws;  // BATCH*HIDDEN*4 = 1.5 MB scratch

  posmlp_gather<<<BATCH, HIDDEN, 0, stream>>>(ids, W1, b1, h);

  // 4096/16 = 256 tiles, one wave each; 8 waves/block -> 32 blocks.
  posmlp_proj_wmma<<<32, 256, 0, stream>>>(h, W2, b2, out);
}
